// _RPN_66013647340247
// MI455X (gfx1250) — compile-verified, matrix operand order checked by reading, not run on hardware
//
#include <hip/hip_runtime.h>
#include <stdint.h>

// ---------------- types ----------------
typedef __bf16 bf16_t;
typedef bf16_t bf16x16 __attribute__((ext_vector_type(16)));
typedef float  f32x8   __attribute__((ext_vector_type(8)));

struct __align__(16) U4 { unsigned x, y, z, w; };
struct __align__(8)  U2 { unsigned x, y; };
union FragAB { bf16x16 v; U4 q[2]; };

// ---------------- problem constants ----------------
#define HID     512
#define KHEAD   8192          // 512*16
#define MTOT    17928         // 540 + 17280 + 36 + 72
#define NHEAD   784           // 4*14*14
#define NCONV   12544         // 4*16*196
#define OFF_CLS    0
#define OFF_BBOX   423360
#define OFF_CLS16  13970880
#define OFF_BBOX16 13999104

// workspace byte offsets
#define WSO_WA    0           // bf16 [27][512][256]        = 7,077,888 B
#define WSO_FEAT  7077888     // bf16 [4][16][196][256]     = 6,422,528 B
#define WSO_ACT   13500416    // bf16 [784][8192]           = 12,845,056 B
#define WSO_CLS   26345472    // f32  [4][540][196]         = 1,693,440 B
#define WSO_CLS16 28038912    // f32  [4][36][196]          = 112,896 B

// ---------------- helpers ----------------
__device__ __forceinline__ unsigned short f2bf(float f) {
  unsigned a = __float_as_uint(f);
  return (unsigned short)((a + 0x7fffu + ((a >> 16) & 1u)) >> 16);
}
__device__ __forceinline__ unsigned pk_bf16(float lo, float hi) {
  unsigned a = __float_as_uint(lo);
  a = (a + 0x7fffu + ((a >> 16) & 1u)) >> 16;
  unsigned b = __float_as_uint(hi);
  b = (b + 0x7fffu + ((b >> 16) & 1u)) & 0xffff0000u;
  return b | a;
}
// async global->LDS copy (ASYNCcnt path, CDNA5 ISA)
__device__ __forceinline__ void async_b128(unsigned lds_off, const void* g) {
  asm volatile("global_load_async_to_lds_b128 %0, %1, off"
               :: "v"(lds_off), "v"((unsigned long long)(uintptr_t)g)
               : "memory");
}
__device__ __forceinline__ void wait_async() {
  asm volatile("s_wait_asynccnt 0x0" ::: "memory");
}

__device__ __forceinline__ const float* head_wrow(int c, const float* cw, const float* bw,
                                                  const float* c16w, const float* b16w) {
  if (c < 540)   return cw   + (size_t)c * KHEAD;
  if (c < 17820) return bw   + (size_t)(c - 540) * KHEAD;
  if (c < 17856) return c16w + (size_t)(c - 17820) * KHEAD;
  if (c < MTOT)  return b16w + (size_t)(c - 17856) * KHEAD;
  return nullptr;
}
__device__ __forceinline__ float head_bias(int c, const float* cb, const float* bb,
                                           const float* c16b, const float* b16b) {
  if (c < 540)   return cb[c];
  if (c < 17820) return bb[c - 540];
  if (c < 17856) return c16b[c - 17820];
  return b16b[c - 17856];
}
__device__ __forceinline__ float* head_dst(int c, int b, int hw, float* out,
                                           float* wscls, float* wscls16) {
  if (c < 540)   return wscls + ((size_t)b * 540 + c) * 196 + hw;
  if (c < 17820) return out + OFF_BBOX + ((size_t)b * 17280 + (c - 540)) * 196 + hw;
  if (c < 17856) return wscls16 + ((size_t)b * 36 + (c - 17820)) * 196 + hw;
  return out + OFF_BBOX16 + ((size_t)b * 72 + (c - 17856)) * 196 + hw;
}

// ---------------- layout transforms (tiny, memory-trivial) ----------------
__global__ void conv_w_xpose(const float* __restrict__ w, unsigned short* __restrict__ wA) {
  int i = blockIdx.x * blockDim.x + threadIdx.x;
  if (i >= 27 * 512 * 256) return;
  int ci = i & 255; int j = i >> 8; int co = j & 511; int tap = j >> 9;
  wA[i] = f2bf(w[((size_t)co * 256 + ci) * 27 + tap]);
}
__global__ void feat_xpose(const float* __restrict__ x, unsigned short* __restrict__ f) {
  int i = blockIdx.x * blockDim.x + threadIdx.x;
  if (i >= 4 * 16 * 196 * 256) return;
  int ci = i & 255; int j = i >> 8;          // j = (b*16+t)*196 + hw
  int hw = j % 196; int k = j / 196; int t = k & 15; int b = k >> 4;
  f[i] = f2bf(x[(((size_t)b * 256 + ci) * 16 + t) * 196 + hw]);
}

// ---------------- conv3d 3x3x3 + ReLU as WMMA implicit GEMM ----------------
// M=512, N=12544, K=27 taps x 256; block tile 64(M) x 128(N), K-step 32.
__global__ void __launch_bounds__(256)
conv3d_wmma(const unsigned short* __restrict__ wA,    // [27][512][256] bf16
            const unsigned short* __restrict__ feat,  // [4][16][196][256] bf16
            const float* __restrict__ conv_b,
            unsigned short* __restrict__ act)         // [784][8192] bf16
{
  __shared__ __align__(16) unsigned lds_a[2][64][16];
  __shared__ __align__(16) unsigned lds_b[2][128][16];
  const int tid  = threadIdx.x;
  const int lane = tid & 31;
  const int wave = tid >> 5;
  const int half = lane >> 4;
  const int l15  = lane & 15;
  const int cBase = blockIdx.y * 64;
  const int nBase = blockIdx.x * 128;
  const int m0 = (wave & 1) * 32;
  const int n0 = (wave >> 1) * 32;

  // per-thread staging coordinates
  const int arow = tid >> 2, achunk = tid & 3;
  const int bchunk = tid & 3;
  int brow[2], bb_[2], bt_[2], bh_[2], bw_[2];
#pragma unroll
  for (int i = 0; i < 2; ++i) {
    int row = (tid >> 2) + i * 64;
    brow[i] = row;
    int n = nBase + row;
    int b = n / 3136; int r1 = n - b * 3136;
    int t = r1 / 196; int hw = r1 - t * 196;
    bb_[i] = b; bt_[i] = t; bh_[i] = hw / 14; bw_[i] = hw - (hw / 14) * 14;
  }

  f32x8 acc[2][2] = {};
  const int KT = 27 * 8;

  auto stage = [&](int kt, int buf) {
    int tap = kt >> 3;
    int ks  = (kt & 7) * 32;
    int dt = tap / 9, rem = tap - dt * 9;
    int dh = rem / 3, dw = rem - dh * 3;
    // A tile 64x32 bf16: async copy
    {
      const unsigned short* g = wA + ((size_t)(tap * 512 + cBase + arow) * 256 + ks) + achunk * 8;
      async_b128((unsigned)(uintptr_t)&lds_a[buf][arow][achunk * 4], g);
    }
    // B tile 128x32 bf16: async copy with zero-fill at padded borders
#pragma unroll
    for (int i = 0; i < 2; ++i) {
      int tp = bt_[i] + dt - 1, hp = bh_[i] + dh - 1, wp = bw_[i] + dw - 1;
      unsigned lo = (unsigned)(uintptr_t)&lds_b[buf][brow[i]][bchunk * 4];
      if ((unsigned)tp < 16u && (unsigned)hp < 14u && (unsigned)wp < 14u) {
        const unsigned short* g = feat +
            ((size_t)((bb_[i] * 16 + tp) * 196 + hp * 14 + wp) * 256 + ks) + bchunk * 8;
        async_b128(lo, g);
      } else {
        *(U4*)(uintptr_t)lo = U4{0u, 0u, 0u, 0u};
      }
    }
  };
  auto compute = [&](int buf) {
    FragAB a0, a1, b0, b1;
    const U4* pa0 = (const U4*)&lds_a[buf][m0 + l15][0];
    const U4* pa1 = (const U4*)&lds_a[buf][m0 + 16 + l15][0];
    const U4* pb0 = (const U4*)&lds_b[buf][n0 + l15][0];
    const U4* pb1 = (const U4*)&lds_b[buf][n0 + 16 + l15][0];
    a0.q[0] = pa0[half]; a0.q[1] = pa0[half + 2];
    a1.q[0] = pa1[half]; a1.q[1] = pa1[half + 2];
    b0.q[0] = pb0[half]; b0.q[1] = pb0[half + 2];
    b1.q[0] = pb1[half]; b1.q[1] = pb1[half + 2];
    acc[0][0] = __builtin_amdgcn_wmma_f32_16x16x32_bf16(false, a0.v, false, b0.v, (short)0, acc[0][0], false, false);
    acc[0][1] = __builtin_amdgcn_wmma_f32_16x16x32_bf16(false, a0.v, false, b1.v, (short)0, acc[0][1], false, false);
    acc[1][0] = __builtin_amdgcn_wmma_f32_16x16x32_bf16(false, a1.v, false, b0.v, (short)0, acc[1][0], false, false);
    acc[1][1] = __builtin_amdgcn_wmma_f32_16x16x32_bf16(false, a1.v, false, b1.v, (short)0, acc[1][1], false, false);
  };

  stage(0, 0);
  for (int kt = 0; kt < KT; ++kt) {
    int buf = kt & 1;
    wait_async();
    __syncthreads();
    if (kt + 1 < KT) stage(kt + 1, buf ^ 1);
    compute(buf);
  }

  // epilogue: bias + ReLU, store bf16 activation [n=b*196+hw][k=co*16+t]
#pragma unroll
  for (int fc = 0; fc < 2; ++fc) {
    int n = nBase + n0 + l15 + fc * 16;
    int b = n / 3136; int r1 = n - b * 3136;
    int t = r1 / 196; int hw = r1 - t * 196;
    unsigned short* dst = act + ((size_t)(b * 196 + hw) * KHEAD + t);
#pragma unroll
    for (int fr = 0; fr < 2; ++fr) {
#pragma unroll
      for (int r = 0; r < 8; ++r) {
        int co = cBase + m0 + fr * 16 + half * 8 + r;
        float v = acc[fr][fc][r] + conv_b[co];
        v = v > 0.f ? v : 0.f;
        dst[(size_t)co * 16] = f2bf(v);
      }
    }
  }
}

// ---------------- all four heads as one WMMA GEMM ----------------
// M=17928 (pad 18048), K=8192, N=784 (pad 832); block 128(M) x 64(N), K-step 32.
__global__ void __launch_bounds__(256)
heads_wmma(const unsigned short* __restrict__ act,
           const float* __restrict__ cls_w, const float* __restrict__ bbox_w,
           const float* __restrict__ cls16_w, const float* __restrict__ bbox16_w,
           const float* __restrict__ cls_b, const float* __restrict__ bbox_b,
           const float* __restrict__ cls16_b, const float* __restrict__ bbox16_b,
           float* __restrict__ out, float* __restrict__ ws_cls, float* __restrict__ ws_cls16)
{
  __shared__ __align__(16) unsigned lds_a[2][128][16];
  __shared__ __align__(16) unsigned lds_b[2][64][16];
  const int tid  = threadIdx.x;
  const int lane = tid & 31;
  const int wave = tid >> 5;
  const int half = lane >> 4;
  const int l15  = lane & 15;
  const int cBase = blockIdx.y * 128;
  const int nBase = blockIdx.x * 64;
  const int m0 = (wave & 3) * 32;
  const int n0 = (wave >> 2) * 32;

  // A: each thread owns one (row, k-half); weight row pointer fixed for the kernel
  const int arow = tid >> 1, ahalf = tid & 1;
  const float* wrow = head_wrow(cBase + arow, cls_w, bbox_w, cls16_w, bbox16_w);
  // B: each thread owns one 16B chunk of one activation row
  const int brow = tid >> 2;
  int bn = nBase + brow; if (bn > NHEAD - 1) bn = NHEAD - 1;
  const unsigned short* bsrc = act + (size_t)bn * KHEAD + (tid & 3) * 8;

  f32x8 acc[2][2] = {};

  auto stage = [&](int k0, int buf) {
    // activations: bf16 already -> async global->LDS
    async_b128((unsigned)(uintptr_t)&lds_b[buf][brow][(tid & 3) * 4], bsrc + k0);
    // weights: fp32 global -> cvt bf16 -> LDS
    U2* dsa = (U2*)&lds_a[buf][arow][ahalf * 8];
    if (wrow) {
      const float4* p = (const float4*)(wrow + k0 + ahalf * 16);
      __builtin_prefetch(wrow + k0 + 256, 0, 0);   // stream weights ahead
#pragma unroll
      for (int j = 0; j < 4; ++j) {
        float4 f = p[j];
        dsa[j] = U2{pk_bf16(f.x, f.y), pk_bf16(f.z, f.w)};
      }
    } else {
#pragma unroll
      for (int j = 0; j < 4; ++j) dsa[j] = U2{0u, 0u};
    }
  };

  stage(0, 0);
  const int KT = KHEAD / 32;   // 256
  for (int kt = 0; kt < KT; ++kt) {
    int buf = kt & 1;
    wait_async();
    __syncthreads();
    if (kt + 1 < KT) stage((kt + 1) * 32, buf ^ 1);
    FragAB a0, a1, b0, b1;
    const U4* pa0 = (const U4*)&lds_a[buf][m0 + l15][0];
    const U4* pa1 = (const U4*)&lds_a[buf][m0 + 16 + l15][0];
    const U4* pb0 = (const U4*)&lds_b[buf][n0 + l15][0];
    const U4* pb1 = (const U4*)&lds_b[buf][n0 + 16 + l15][0];
    a0.q[0] = pa0[half]; a0.q[1] = pa0[half + 2];
    a1.q[0] = pa1[half]; a1.q[1] = pa1[half + 2];
    b0.q[0] = pb0[half]; b0.q[1] = pb0[half + 2];
    b1.q[0] = pb1[half]; b1.q[1] = pb1[half + 2];
    acc[0][0] = __builtin_amdgcn_wmma_f32_16x16x32_bf16(false, a0.v, false, b0.v, (short)0, acc[0][0], false, false);
    acc[0][1] = __builtin_amdgcn_wmma_f32_16x16x32_bf16(false, a0.v, false, b1.v, (short)0, acc[0][1], false, false);
    acc[1][0] = __builtin_amdgcn_wmma_f32_16x16x32_bf16(false, a1.v, false, b0.v, (short)0, acc[1][0], false, false);
    acc[1][1] = __builtin_amdgcn_wmma_f32_16x16x32_bf16(false, a1.v, false, b1.v, (short)0, acc[1][1], false, false);
  }

  // epilogue: bias add, scatter to head outputs (cls raw scores go to ws)
#pragma unroll
  for (int fc = 0; fc < 2; ++fc) {
    int n = nBase + n0 + l15 + fc * 16;
    if (n >= NHEAD) continue;
    int b = n / 196; int hw = n - b * 196;
#pragma unroll
    for (int fr = 0; fr < 2; ++fr) {
#pragma unroll
      for (int r = 0; r < 8; ++r) {
        int c = cBase + m0 + fr * 16 + half * 8 + r;
        if (c >= MTOT) continue;
        float v = acc[fr][fc][r] + head_bias(c, cls_b, bbox_b, cls16_b, bbox16_b);
        *head_dst(c, b, hw, out, ws_cls, ws_cls16) = v;
      }
    }
  }
}

// ---------------- pair softmax: (c, c+270) and (c, c+18) ----------------
__global__ void softmax_pairs(const float* __restrict__ wc, const float* __restrict__ wc16,
                              float* __restrict__ out) {
  int i = blockIdx.x * blockDim.x + threadIdx.x;
  const int N1 = 4 * 270 * 196;
  const int N2 = 4 * 18 * 196;
  if (i < N1) {
    int hw = i % 196; int j = i / 196; int c = j % 270; int b = j / 270;
    size_t i0 = ((size_t)b * 540 + c) * 196 + hw;
    size_t i1 = ((size_t)b * 540 + c + 270) * 196 + hw;
    float s0 = wc[i0], s1 = wc[i1];
    float m = fmaxf(s0, s1);
    float e0 = __expf(s0 - m), e1 = __expf(s1 - m);
    float inv = 1.0f / (e0 + e1);
    out[OFF_CLS + i0] = e0 * inv;
    out[OFF_CLS + i1] = e1 * inv;
  } else if (i < N1 + N2) {
    int q = i - N1;
    int hw = q % 196; int j = q / 196; int c = j % 18; int b = j / 18;
    size_t i0 = ((size_t)b * 36 + c) * 196 + hw;
    size_t i1 = ((size_t)b * 36 + c + 18) * 196 + hw;
    float s0 = wc16[i0], s1 = wc16[i1];
    float m = fmaxf(s0, s1);
    float e0 = __expf(s0 - m), e1 = __expf(s1 - m);
    float inv = 1.0f / (e0 + e1);
    out[OFF_CLS16 + i0] = e0 * inv;
    out[OFF_CLS16 + i1] = e1 * inv;
  }
}

// ---------------- launcher ----------------
extern "C" void kernel_launch(void* const* d_in, const int* in_sizes, int n_in,
                              void* d_out, int out_size, void* d_ws, size_t ws_size,
                              hipStream_t stream) {
  (void)in_sizes; (void)n_in; (void)out_size; (void)ws_size;
  const float* base_feat = (const float*)d_in[0];
  const float* conv_w    = (const float*)d_in[4];
  const float* conv_b    = (const float*)d_in[5];
  const float* cls_w     = (const float*)d_in[6];
  const float* cls_b     = (const float*)d_in[7];
  const float* bbox_w    = (const float*)d_in[8];
  const float* bbox_b    = (const float*)d_in[9];
  const float* cls16_w   = (const float*)d_in[10];
  const float* cls16_b   = (const float*)d_in[11];
  const float* bbox16_w  = (const float*)d_in[12];
  const float* bbox16_b  = (const float*)d_in[13];

  char* ws = (char*)d_ws;
  unsigned short* wA     = (unsigned short*)(ws + WSO_WA);
  unsigned short* feat   = (unsigned short*)(ws + WSO_FEAT);
  unsigned short* act    = (unsigned short*)(ws + WSO_ACT);
  float* ws_cls          = (float*)(ws + WSO_CLS);
  float* ws_cls16        = (float*)(ws + WSO_CLS16);
  float* out             = (float*)d_out;

  conv_w_xpose<<<(27 * 512 * 256 + 255) / 256, 256, 0, stream>>>(conv_w, wA);
  feat_xpose<<<(4 * 16 * 196 * 256 + 255) / 256, 256, 0, stream>>>(base_feat, feat);
  conv3d_wmma<<<dim3(98, 8), 256, 0, stream>>>(wA, feat, conv_b, act);
  heads_wmma<<<dim3(13, 141), 256, 0, stream>>>(act, cls_w, bbox_w, cls16_w, bbox16_w,
                                                cls_b, bbox_b, cls16_b, bbox16_b,
                                                out, ws_cls, ws_cls16);
  softmax_pairs<<<(4 * 288 * 196 + 255) / 256, 256, 0, stream>>>(ws_cls, ws_cls16, out);
}
